// LightAGSM_33663953666140
// MI455X (gfx1250) — matrix operand order repaired and checked by reading, not compiled, not run on hardware
//
#include <hip/hip_runtime.h>
#include <math.h>

#define N_NODES 16384
#define IN_DIM  256
#define HID     128
#define KNN     16
#define E_EDGES (N_NODES * KNN)
#define POS_TILE 512

typedef _Float16 f16;
typedef _Float16 v16h __attribute__((ext_vector_type(16)));
typedef _Float16 h8   __attribute__((ext_vector_type(8)));
typedef float    v8f  __attribute__((ext_vector_type(8)));

// ---------------------------------------------------------------------------
// kNN: one thread per node, register-resident top-16 (ascending d2, ties ->
// lower index first, matching jax.lax.top_k on -d2). Candidate positions are
// staged through LDS in 512-node tiles (broadcast ds reads in the inner loop).
// Same distance formula as the reference: |a|^2 + |b|^2 - 2 a.b
// ---------------------------------------------------------------------------
__global__ void __launch_bounds__(256) knn_kernel(const float* __restrict__ pos,
                                                  int* __restrict__ knn_idx) {
    __shared__ float2 spos[POS_TILE];
    const float2* pos2 = reinterpret_cast<const float2*>(pos);

    int i = blockIdx.x * blockDim.x + threadIdx.x;  // grid is exact: N/256 blocks
    float2 pi = pos2[i];
    float sqi = pi.x * pi.x + pi.y * pi.y;

    float d[KNN];
    int   id[KNN];
#pragma unroll
    for (int p = 0; p < KNN; ++p) { d[p] = 3.4e38f; id[p] = -1; }

    for (int base = 0; base < N_NODES; base += POS_TILE) {
        // cooperative stage of this position tile into LDS
#pragma unroll
        for (int t = 0; t < POS_TILE / 256; ++t)
            spos[threadIdx.x + t * 256] = pos2[base + threadIdx.x + t * 256];
        __syncthreads();

        for (int jj = 0; jj < POS_TILE; ++jj) {
            int j = base + jj;
            if (j == i) continue;
            float2 pj = spos[jj];
            float sqj = pj.x * pj.x + pj.y * pj.y;
            float dd = sqi + sqj - 2.0f * (pi.x * pj.x + pi.y * pj.y);
            if (dd < d[KNN - 1]) {
                int cand = j;
#pragma unroll
                for (int p = 0; p < KNN; ++p) {
                    if (dd < d[p]) {
                        float td = d[p]; d[p] = dd; dd = td;
                        int   ti = id[p]; id[p] = cand; cand = ti;
                    }
                }
            }
        }
        __syncthreads();
    }
#pragma unroll
    for (int p = 0; p < KNN; ++p) knn_idx[i * KNN + p] = id[p];
}

// ---------------------------------------------------------------------------
// Mutual mask: edge t = (i, j) is mutual iff i appears in knn(j).
// ---------------------------------------------------------------------------
__global__ void __launch_bounds__(256) mutual_kernel(const int* __restrict__ knn_idx,
                                                     float* __restrict__ maskE) {
    int t = blockIdx.x * blockDim.x + threadIdx.x;
    if (t >= E_EDGES) return;
    int i = t >> 4;
    int j = knn_idx[t];
    float m = 0.0f;
#pragma unroll
    for (int s = 0; s < KNN; ++s)
        if (knn_idx[j * KNN + s] == i) m = 1.0f;
    maskE[t] = m;
}

// ---------------------------------------------------------------------------
// fp32 -> fp16 conversion
// ---------------------------------------------------------------------------
__global__ void __launch_bounds__(256) cvt_f32_f16(const float* __restrict__ in,
                                                   f16* __restrict__ out, int n) {
    int i = blockIdx.x * blockDim.x + threadIdx.x;
    if (i < n) out[i] = (f16)in[i];
}

// ---------------------------------------------------------------------------
// Weight prep: build transposed, fp16, concatenated weights.
// Layer 1: Wcat1T [512][256] rows = {q,k,v,r} columns; biascat1[512] (r -> 0)
// ---------------------------------------------------------------------------
__global__ void __launch_bounds__(256) prep_w1(const float* __restrict__ Wq,
                                               const float* __restrict__ Wk,
                                               const float* __restrict__ Wv,
                                               const float* __restrict__ Wr,
                                               const float* __restrict__ bq,
                                               const float* __restrict__ bk,
                                               const float* __restrict__ bv,
                                               f16* __restrict__ wt,
                                               float* __restrict__ bc) {
    int t = blockIdx.x * blockDim.x + threadIdx.x;  // 512*256 threads
    if (t >= 512 * IN_DIM) return;
    int r = t / IN_DIM, k = t % IN_DIM;
    int sel = r >> 7, n = r & 127;
    const float* W = (sel == 0) ? Wq : (sel == 1) ? Wk : (sel == 2) ? Wv : Wr;
    wt[(size_t)r * IN_DIM + k] = (f16)W[k * HID + n];
    if (k == 0) {
        float b = (sel == 0) ? bq[n] : (sel == 1) ? bk[n] : (sel == 2) ? bv[n] : 0.0f;
        bc[r] = b;
    }
}

// Layer 2: Wcat2T [384][128] rows = {q,k,v}; biascat2[384]
__global__ void __launch_bounds__(256) prep_w2(const float* __restrict__ Wq,
                                               const float* __restrict__ Wk,
                                               const float* __restrict__ Wv,
                                               const float* __restrict__ bq,
                                               const float* __restrict__ bk,
                                               const float* __restrict__ bv,
                                               f16* __restrict__ wt,
                                               float* __restrict__ bc) {
    int t = blockIdx.x * blockDim.x + threadIdx.x;  // 384*128 threads
    if (t >= 384 * HID) return;
    int r = t / HID, k = t % HID;
    int sel = r >> 7, n = r & 127;
    const float* W = (sel == 0) ? Wq : (sel == 1) ? Wk : Wv;
    wt[(size_t)r * HID + k] = (f16)W[k * HID + n];
    if (k == 0) {
        float b = (sel == 0) ? bq[n] : (sel == 1) ? bk[n] : bv[n];
        bc[r] = b;
    }
}

// ---------------------------------------------------------------------------
// Fragment loaders per CDNA5 ISA 7.12.2 wave32 layouts.
//   A 16x32 f16: lane<16 holds row (lane&15), K = {kc+0..7, kc+16..23};
//                lane>=16 same row, K = {kc+8..15, kc+24..31}
//   B 32x16 f16: lane holds col (lane&15), K same as A: {kc+8*hg+0..7, kc+16+8*hg+0..7}
// ---------------------------------------------------------------------------
__device__ __forceinline__ v16h load_frag_a(const f16* __restrict__ p, int kc, int hg) {
    h8 x0 = *reinterpret_cast<const h8*>(p + kc + 8 * hg);
    h8 x1 = *reinterpret_cast<const h8*>(p + kc + 16 + 8 * hg);
    v16h r;
#pragma unroll
    for (int q = 0; q < 8; ++q) { r[q] = x0[q]; r[8 + q] = x1[q]; }
    return r;
}

__device__ __forceinline__ v16h load_frag_b(const f16* __restrict__ p, int kc, int hg) {
    h8 x0 = *reinterpret_cast<const h8*>(p + kc + 8 * hg);
    h8 x1 = *reinterpret_cast<const h8*>(p + kc + 16 + 8 * hg);
    v16h r;
#pragma unroll
    for (int q = 0; q < 8; ++q) { r[q] = x0[q]; r[8 + q] = x1[q]; }
    return r;
}

// ---------------------------------------------------------------------------
// WMMA GEMM: C[M,Nout] = A[M,Kd](f16) * Bt[Nout,Kd]^T(f16) + bias, f32 accum.
// One wave32 per 32x32 output tile: 2x2 WMMA accumulators, each A/B fragment
// reused twice -> 4 v_wmma per 8 b128 loads (2x the arithmetic intensity of
// a 16x16-per-wave scheme).
//   D 16x16 f32: acc[r] = C[tile_m + r + 8*(lane>=16)][tile_n + (lane&15)]
// ---------------------------------------------------------------------------
__global__ void __launch_bounds__(128) gemm_wmma(const f16* __restrict__ A,
                                                 const f16* __restrict__ Bt,
                                                 const float* __restrict__ bias,
                                                 float* __restrict__ C,
                                                 int Kd, int Nout) {
    int wid  = blockIdx.x * 4 + (threadIdx.x >> 5);
    int lane = threadIdx.x & 31;
    int n2   = Nout >> 5;         // number of 32-col tiles
    int tm   = wid / n2;          // 32-row tile index
    int tn   = wid % n2;          // 32-col tile index
    int hg   = lane >> 4;
    int l15  = lane & 15;

    const f16* Ap0 = A  + (size_t)(tm * 32 + l15) * Kd;
    const f16* Ap1 = Ap0 + (size_t)16 * Kd;
    const f16* Bp0 = Bt + (size_t)(tn * 32 + l15) * Kd;
    const f16* Bp1 = Bp0 + (size_t)16 * Kd;

    v8f acc00 = {}, acc01 = {}, acc10 = {}, acc11 = {};
    for (int kc = 0; kc < Kd; kc += 32) {
        v16h a0 = load_frag_a(Ap0, kc, hg);
        v16h a1 = load_frag_a(Ap1, kc, hg);
        v16h b0 = load_frag_b(Bp0, kc, hg);
        v16h b1 = load_frag_b(Bp1, kc, hg);
        acc00 = __builtin_amdgcn_wmma_f32_16x16x32_f16(false, a0, false, b0, (short)0,
                                                       acc00, false, false);
        acc01 = __builtin_amdgcn_wmma_f32_16x16x32_f16(false, a0, false, b1, (short)0,
                                                       acc01, false, false);
        acc10 = __builtin_amdgcn_wmma_f32_16x16x32_f16(false, a1, false, b0, (short)0,
                                                       acc10, false, false);
        acc11 = __builtin_amdgcn_wmma_f32_16x16x32_f16(false, a1, false, b1, (short)0,
                                                       acc11, false, false);
    }

    int ncol0 = tn * 32 + l15;
    int ncol1 = ncol0 + 16;
    float bv0 = bias ? bias[ncol0] : 0.0f;
    float bv1 = bias ? bias[ncol1] : 0.0f;
#pragma unroll
    for (int r = 0; r < 8; ++r) {
        int crow0 = tm * 32 + r + 8 * hg;
        int crow1 = crow0 + 16;
        C[(size_t)crow0 * Nout + ncol0] = acc00[r] + bv0;
        C[(size_t)crow0 * Nout + ncol1] = acc01[r] + bv1;
        C[(size_t)crow1 * Nout + ncol0] = acc10[r] + bv0;
        C[(size_t)crow1 * Nout + ncol1] = acc11[r] + bv1;
    }
}

// ---------------------------------------------------------------------------
// Edge gate: e = dot(q[row], k[col]) / sqrt(HID); g = sigmoid((e-tau)/0.7)*mask
// ---------------------------------------------------------------------------
__global__ void __launch_bounds__(256) edge_gate(const float* __restrict__ Q,
                                                 const float* __restrict__ Km,
                                                 int stride,
                                                 const int* __restrict__ col,
                                                 const float* __restrict__ maskE,
                                                 const float* __restrict__ tauPtr,
                                                 float* __restrict__ g) {
    int t = blockIdx.x * blockDim.x + threadIdx.x;
    if (t >= E_EDGES) return;
    int i = t >> 4;
    int j = col[t];
    const float4* q = reinterpret_cast<const float4*>(Q + (size_t)i * stride);
    const float4* k = reinterpret_cast<const float4*>(Km + (size_t)j * stride);
    float e = 0.0f;
#pragma unroll
    for (int h = 0; h < HID / 4; ++h) {
        float4 a = q[h], b = k[h];
        e += a.x * b.x + a.y * b.y + a.z * b.z + a.w * b.w;
    }
    e *= 0.08838834764831845f;  // 1/sqrt(128)
    float z = (e - tauPtr[0]) * (1.0f / 0.7f);
    float s = 1.0f / (1.0f + expf(-z));
    g[t] = s * maskE[t];
}

// ---------------------------------------------------------------------------
// Per-node aggregation + residual + LayerNorm + ReLU. One wave32 per node,
// 4 feature dims per lane (dim = c*32 + lane), shfl_xor tree reductions.
// ---------------------------------------------------------------------------
__global__ void __launch_bounds__(256) agg_ln(const float* __restrict__ V, int vstride,
                                              const float* __restrict__ Res, int rstride,
                                              const int* __restrict__ col,
                                              const float* __restrict__ g,
                                              const float* __restrict__ gamma,
                                              const float* __restrict__ beta,
                                              float* __restrict__ outF,
                                              f16* __restrict__ outH) {
    int node = blockIdx.x * 8 + (threadIdx.x >> 5);
    int lane = threadIdx.x & 31;

    // per-row normalizer: den = max(sum g, 1e-6)
    float den = (lane < KNN) ? g[node * KNN + lane] : 0.0f;
#pragma unroll
    for (int o = 16; o > 0; o >>= 1) den += __shfl_xor(den, o, 32);
    den = fmaxf(den, 1e-6f);

    float acc[4];
#pragma unroll
    for (int c = 0; c < 4; ++c)
        acc[c] = Res[(size_t)node * rstride + c * 32 + lane];

    for (int e = 0; e < KNN; ++e) {
        float ge = g[node * KNN + e];
        if (ge == 0.0f) continue;
        float alpha = ge / den;
        int j = col[node * KNN + e];
        const float* vp = V + (size_t)j * vstride;
#pragma unroll
        for (int c = 0; c < 4; ++c)
            acc[c] += alpha * vp[c * 32 + lane];
    }

    float s = 0.0f, s2 = 0.0f;
#pragma unroll
    for (int c = 0; c < 4; ++c) { s += acc[c]; s2 += acc[c] * acc[c]; }
#pragma unroll
    for (int o = 16; o > 0; o >>= 1) {
        s  += __shfl_xor(s,  o, 32);
        s2 += __shfl_xor(s2, o, 32);
    }
    float mu   = s * (1.0f / HID);
    float var  = s2 * (1.0f / HID) - mu * mu;
    float rinv = rsqrtf(var + 1e-5f);

#pragma unroll
    for (int c = 0; c < 4; ++c) {
        int h = c * 32 + lane;
        float o = gamma[h] * (acc[c] - mu) * rinv + beta[h];
        o = fmaxf(o, 0.0f);
        outF[(size_t)node * HID + h] = o;
        if (outH) outH[(size_t)node * HID + h] = (f16)o;
    }
}

// ---------------------------------------------------------------------------
extern "C" void kernel_launch(void* const* d_in, const int* in_sizes, int n_in,
                              void* d_out, int out_size, void* d_ws, size_t ws_size,
                              hipStream_t stream) {
    const float* x      = (const float*)d_in[0];
    const float* pos    = (const float*)d_in[1];
    const float* Wq1    = (const float*)d_in[2];
    const float* bq1    = (const float*)d_in[3];
    const float* Wk1    = (const float*)d_in[4];
    const float* bk1    = (const float*)d_in[5];
    const float* Wv1    = (const float*)d_in[6];
    const float* bv1    = (const float*)d_in[7];
    const float* Wr1    = (const float*)d_in[8];
    const float* gamma1 = (const float*)d_in[9];
    const float* beta1  = (const float*)d_in[10];
    const float* Wq2    = (const float*)d_in[11];
    const float* bq2    = (const float*)d_in[12];
    const float* Wk2    = (const float*)d_in[13];
    const float* bk2    = (const float*)d_in[14];
    const float* Wv2    = (const float*)d_in[15];
    const float* bv2    = (const float*)d_in[16];
    const float* gamma2 = (const float*)d_in[17];
    const float* beta2  = (const float*)d_in[18];
    const float* tau    = (const float*)d_in[19];

    float* out = (float*)d_out;
    float* x2  = out;                                // [N*HID]
    float* g1  = out + (size_t)N_NODES * HID;        // [E]
    float* g2  = g1 + E_EDGES;                       // [E]

    // carve workspace (256B-aligned blocks)
    char* w = (char*)d_ws;
    auto alloc = [&](size_t bytes) -> char* {
        char* p = w;
        w += (bytes + 255) & ~(size_t)255;
        return p;
    };
    int*   knn_idx = (int*)  alloc((size_t)E_EDGES * 4);
    float* maskE   = (float*)alloc((size_t)E_EDGES * 4);
    f16*   xh      = (f16*)  alloc((size_t)N_NODES * IN_DIM * 2);
    f16*   w1t     = (f16*)  alloc((size_t)512 * IN_DIM * 2);
    float* b1c     = (float*)alloc(512 * 4);
    float* qkvr1   = (float*)alloc((size_t)N_NODES * 512 * 4);
    float* x1      = (float*)alloc((size_t)N_NODES * HID * 4);
    f16*   x1h     = (f16*)  alloc((size_t)N_NODES * HID * 2);
    f16*   w2t     = (f16*)  alloc((size_t)384 * HID * 2);
    float* b2c     = (float*)alloc(384 * 4);
    float* qkv2    = (float*)alloc((size_t)N_NODES * 384 * 4);
    (void)ws_size; (void)in_sizes; (void)n_in; (void)out_size;

    // Graph construction
    knn_kernel<<<N_NODES / 256, 256, 0, stream>>>(pos, knn_idx);
    mutual_kernel<<<E_EDGES / 256, 256, 0, stream>>>(knn_idx, maskE);

    // Layer 1: fused q|k|v|res GEMM [16384,256] x [256,512]
    cvt_f32_f16<<<(N_NODES * IN_DIM) / 256, 256, 0, stream>>>(x, xh, N_NODES * IN_DIM);
    prep_w1<<<(512 * IN_DIM) / 256, 256, 0, stream>>>(Wq1, Wk1, Wv1, Wr1, bq1, bk1, bv1,
                                                      w1t, b1c);
    gemm_wmma<<<(N_NODES / 32) * (512 / 32) / 4, 128, 0, stream>>>(xh, w1t, b1c, qkvr1,
                                                                   IN_DIM, 512);
    edge_gate<<<E_EDGES / 256, 256, 0, stream>>>(qkvr1 + 0, qkvr1 + HID, 512,
                                                 knn_idx, maskE, tau, g1);
    agg_ln<<<N_NODES / 8, 256, 0, stream>>>(qkvr1 + 2 * HID, 512,   // V
                                            qkvr1 + 3 * HID, 512,   // res
                                            knn_idx, g1, gamma1, beta1, x1, x1h);

    // Layer 2: fused q|k|v GEMM [16384,128] x [128,384], res = identity
    prep_w2<<<(384 * HID) / 256, 256, 0, stream>>>(Wq2, Wk2, Wv2, bq2, bk2, bv2, w2t, b2c);
    gemm_wmma<<<(N_NODES / 32) * (384 / 32) / 4, 128, 0, stream>>>(x1h, w2t, b2c, qkv2,
                                                                   HID, 384);
    edge_gate<<<E_EDGES / 256, 256, 0, stream>>>(qkv2 + 0, qkv2 + HID, 384,
                                                 knn_idx, maskE, tau, g2);
    agg_ln<<<N_NODES / 8, 256, 0, stream>>>(qkv2 + 2 * HID, 384,    // V
                                            x1, HID,                // res = x1
                                            knn_idx, g2, gamma2, beta2, x2, nullptr);
}
